// HeisenbergAttention_24507083391400
// MI455X (gfx1250) — matrix operand order repaired and checked by reading, not compiled, not run on hardware
//
#include <hip/hip_runtime.h>
#include <hip/hip_bf16.h>

// HeisenbergAttention for MI455X (gfx1250, wave32, WMMA + TDM).
// Chunked linear-attention: o_t = q_t @ M_t, M_t = sum_{s<=t} a_excl_s (x) b_s.
// bf16 operands / f32 accumulation via v_wmma_f32_16x16x32_bf16.
// GEMM tiles are DMA'd into LDS with TENSOR_LOAD_TO_LDS (double-buffered,
// TENSORcnt-tracked) so global->LDS movement overlaps the WMMA pipe.

#define B_   2
#define T_   2048
#define DM_  1024
#define H_   16
#define D_   64
#define BT_  (B_ * T_)   // 4096

typedef __bf16 bf16;
typedef __attribute__((ext_vector_type(16))) bf16  v16bf;
typedef __attribute__((ext_vector_type(8)))  float v8f;
typedef __attribute__((ext_vector_type(4)))  unsigned int u32x4;
typedef __attribute__((ext_vector_type(8)))  int i32x8;
typedef __attribute__((ext_vector_type(4)))  int i32x4;

#define WMMA_BF16(a, b, c) \
  __builtin_amdgcn_wmma_f32_16x16x32_bf16(false, (a), false, (b), (short)0, (c), false, false)

static __device__ __forceinline__ bf16 f2bf(float f) {
  union { float f; unsigned int u; } v; v.f = f;
  unsigned int u = v.u + 0x7fffu + ((v.u >> 16) & 1u);   // round-to-nearest-even
  unsigned short s = (unsigned short)(u >> 16);
  return __builtin_bit_cast(bf16, s);
}

// ---- TDM: 2D bf16 tile (tile_k x tile_rows, row stride = stride_elems) ----
// Builds a Tensor DMA Descriptor (ISA 8.3/8.4) and issues TENSOR_LOAD_TO_LDS.
// Issued once per wave (EXEC is ignored by TDM); completion via TENSORcnt.
static __device__ __forceinline__ void tdm_load_tile_bf16(
    unsigned lds_addr, const bf16* gptr,
    unsigned tile_k, unsigned tile_rows, unsigned stride_elems) {
  unsigned long long ga = (unsigned long long)(const void*)gptr;
  u32x4 g0;
  g0[0] = 1u;                                     // count=1 (valid), user mode
  g0[1] = lds_addr;                               // lds_addr   (bits 63:32)
  g0[2] = (unsigned)(ga & 0xffffffffu);           // global_addr[31:0]
  g0[3] = (unsigned)((ga >> 32) & 0x01ffffffu)    // global_addr[56:32]
        | (2u << 30);                             // type=2 ("image")
  const unsigned td0 = stride_elems;              // tensor_dim0 (elements)
  const unsigned td1 = 1u << 20;                  // tensor_dim1 (large, no clip)
  i32x8 g1;
  g1[0] = (int)(1u << 16);                        // data_size=1 -> 2 bytes
  g1[1] = (int)((td0 & 0xffffu) << 16);           // tensor_dim0[15:0] @ bits 63:48
  g1[2] = (int)((td0 >> 16) | ((td1 & 0xffffu) << 16)); // td0[31:16] | td1[15:0]
  g1[3] = (int)((td1 >> 16) | (tile_k << 16));    // td1[31:16] | tile_dim0
  g1[4] = (int)(tile_rows & 0xffffu);             // tile_dim1 (tile_dim2=0)
  g1[5] = (int)stride_elems;                      // tensor_dim0_stride[31:0]
  g1[6] = 0;                                      // stride[47:32] | dim1_stride[15:0]
  g1[7] = 0;                                      // dim1_stride[47:16]
  i32x4 gz = (i32x4){0, 0, 0, 0};                 // groups 2/3 unused (2D tile)
#if defined(__clang_major__) && (__clang_major__ >= 23)
  i32x8 gz8 = (i32x8){0, 0, 0, 0, 0, 0, 0, 0};
  __builtin_amdgcn_tensor_load_to_lds(g0, g1, gz, gz, gz8, 0);
#else
  __builtin_amdgcn_tensor_load_to_lds(g0, g1, gz, gz, 0);
#endif
}

// ---- WMMA fragment gathers (contiguous-k order shared by A and B, wave32) -------------------
// A fragment (16x32, 16-bit), source row-major [m][k], k contiguous:
//   lane = M, element e holds K = (lane>>4)*16 + e
//   -> one contiguous 32B run per lane => ds_load_b128 x2
static __device__ __forceinline__ v16bf frag_A(const bf16* src, int ld, int lane) {
  const int m = lane & 15, half = lane >> 4;
  v16bf f;
#pragma unroll
  for (int e = 0; e < 16; ++e)
    f[e] = src[m * ld + ((half << 4) | e)];
  return f;
}
// B fragment (32x16, 16-bit), source n-major [n][k], k contiguous:
//   lane = N; lanes 0-15 hold K=k0..k0+15, lanes 16-31 hold K=k0+16..k0+31
static __device__ __forceinline__ v16bf frag_Bk(const bf16* src, int ld, int n0, int k0, int lane) {
  const bf16* p = src + (size_t)(n0 + (lane & 15)) * ld + k0 + ((lane >> 4) << 4);
  v16bf f;
#pragma unroll
  for (int e = 0; e < 16; ++e) f[e] = p[e];
  return f;
}
// B fragment from an f32 n-major source (running state M^T), convert on the fly
static __device__ __forceinline__ v16bf frag_Bk_f32(const float* src, int ld, int n0, int k0, int lane) {
  const float* p = src + (size_t)(n0 + (lane & 15)) * ld + k0 + ((lane >> 4) << 4);
  v16bf f;
#pragma unroll
  for (int e = 0; e < 16; ++e) f[e] = f2bf(p[e]);
  return f;
}

// ---- f32 -> bf16 conversion (row-major copy) -------------------------------
__global__ __launch_bounds__(256) void cvt_f32_bf16(const float* __restrict__ in,
                                                    bf16* __restrict__ out, int n) {
  int i = (blockIdx.x * 256 + threadIdx.x) * 4;
  if (i + 4 <= n) {
    float4 v = *(const float4*)(in + i);
    out[i] = f2bf(v.x); out[i + 1] = f2bf(v.y);
    out[i + 2] = f2bf(v.z); out[i + 3] = f2bf(v.w);
  } else {
    for (; i < n; ++i) out[i] = f2bf(in[i]);
  }
}

// ---- f32 (KxN) -> bf16 transposed (NxK): weights stored W^T ---------------
__global__ __launch_bounds__(256) void cvt_f32_bf16_t(const float* __restrict__ in,
                                                      bf16* __restrict__ out,
                                                      int K, int N) {
  int i = blockIdx.x * 256 + threadIdx.x;      // output index, N x K
  if (i < K * N) {
    int n = i / K, k = i - n * K;
    out[i] = f2bf(in[(size_t)k * N + n]);
  }
}

// ---- bf16 GEMM, f32 accumulate: C(MxN) = A(MxK) @ Bt(NxK)^T ---------------
// Block = 256 threads (8 waves). Tile 128(M) x 64(N), K-step 32 (one WMMA).
// Tiles DMA'd into double-buffered LDS by wave 0 via TDM; WMMA overlaps DMA.
#define GTM 128
#define GTN 64
#define GTK 32
__global__ __launch_bounds__(256) void gemm_bf16f32(const bf16* __restrict__ A,
                                                    const bf16* __restrict__ Bt,  // N x K
                                                    float* __restrict__ C,
                                                    int M, int N, int K) {
  __shared__ __align__(16) bf16 As [2][GTM * GTK];   // [m][k]
  __shared__ __align__(16) bf16 BsT[2][GTN * GTK];   // [n][k]
  const int tid = threadIdx.x, lane = tid & 31, wave = tid >> 5;
  const int m0 = blockIdx.y * GTM, n0 = blockIdx.x * GTN;
  const int nk = K / GTK;

  v8f acc[4];
#pragma unroll
  for (int c = 0; c < 4; ++c) acc[c] = (v8f){0.f,0.f,0.f,0.f,0.f,0.f,0.f,0.f};

  const bf16* aTile = A  + (size_t)m0 * K;
  const bf16* bTile = Bt + (size_t)n0 * K;
  const unsigned ldsA0 = (unsigned)(unsigned long long)(void*)&As[0][0];
  const unsigned ldsA1 = (unsigned)(unsigned long long)(void*)&As[1][0];
  const unsigned ldsB0 = (unsigned)(unsigned long long)(void*)&BsT[0][0];
  const unsigned ldsB1 = (unsigned)(unsigned long long)(void*)&BsT[1][0];

  if (wave == 0) {                                   // prologue: tile 0
    tdm_load_tile_bf16(ldsA0, aTile, GTK, GTM, K);
    tdm_load_tile_bf16(ldsB0, bTile, GTK, GTN, K);
  }

  for (int kk = 0; kk < nk; ++kk) {
    const int cur = kk & 1;
    if (wave == 0) {
      if (kk + 1 < nk) {                             // prefetch next tile
        tdm_load_tile_bf16(cur ? ldsA0 : ldsA1, aTile + (kk + 1) * GTK, GTK, GTM, K);
        tdm_load_tile_bf16(cur ? ldsB0 : ldsB1, bTile + (kk + 1) * GTK, GTK, GTN, K);
        __builtin_amdgcn_s_wait_tensorcnt(2);        // current tile's 2 DMAs done
      } else {
        __builtin_amdgcn_s_wait_tensorcnt(0);
      }
    }
    __syncthreads();                                 // all waves see current tile

    v16bf af = frag_A(&As[cur][(wave * 16) * GTK], GTK, lane);
#pragma unroll
    for (int c = 0; c < 4; ++c) {
      v16bf bf = frag_Bk(&BsT[cur][0], GTK, c * 16, 0, lane);
      acc[c] = WMMA_BF16(af, bf, acc[c]);
    }
    __syncthreads();                                 // done reading; buffer reusable
  }

  const int nn = lane & 15, mb = (lane >> 4) << 3;
#pragma unroll
  for (int c = 0; c < 4; ++c)
#pragma unroll
    for (int r = 0; r < 8; ++r)
      C[(size_t)(m0 + wave * 16 + mb + r) * N + (n0 + c * 16 + nn)] = acc[c][r];
}

// ---- per-head LayerNorm over D=64, one wave per row -----------------------
// in:  (B*T, H*D) f32 (GEMM output).  out: (B,H,T,D) f32 (split_heads applied)
__global__ __launch_bounds__(256) void ln_split(const float* __restrict__ y,
                                                const float* __restrict__ gamma,
                                                const float* __restrict__ beta,
                                                float* __restrict__ out) {
  const int lane = threadIdx.x & 31, wave = threadIdx.x >> 5;
  const int row = blockIdx.x * 8 + wave;               // row in [0, B*T*H)
  const float* src = y + (size_t)row * D_;
  float x0 = src[lane], x1 = src[lane + 32];
  float s = x0 + x1, ss = x0 * x0 + x1 * x1;
#pragma unroll
  for (int m = 16; m >= 1; m >>= 1) {
    s  += __shfl_xor(s, m, 32);
    ss += __shfl_xor(ss, m, 32);
  }
  const float mean = s * (1.0f / 64.0f);
  const float var  = ss * (1.0f / 64.0f) - mean * mean;
  const float rs   = rsqrtf(var + 1e-5f);
  const int h = row & (H_ - 1);
  const int bt = row >> 4;                              // b*T + t
  float* dst = out + (((size_t)((bt / T_) * H_ + h) * T_) + (bt % T_)) * D_;
  dst[lane]      = (x0 - mean) * rs * gamma[lane] + beta[lane];
  dst[lane + 32] = (x1 - mean) * rs * gamma[lane + 32] + beta[lane + 32];
}

// ---- chunked linear attention: one block per (b,h), 32 chunks of 64 -------
__global__ __launch_bounds__(256) void attn_kernel(const float* __restrict__ qln,
                                                   const float* __restrict__ aln,
                                                   const float* __restrict__ bln,
                                                   bf16* __restrict__ o) {
  __shared__ __align__(16) float sMt  [D_ * D_];   // state, TRANSPOSED: [n=d_b][m=d_a]
  __shared__ __align__(16) bf16  sQ   [64 * D_];   // [t][d]   (A operand)
  __shared__ __align__(16) bf16  sAex [64 * D_];   // [s][d]   (B op for q@aex^T)
  __shared__ __align__(16) bf16  sAexT[D_ * 64];   // [d][s]   (A op for aex^T@b)
  __shared__ __align__(16) bf16  sBt  [D_ * 64];   // [d][s]   (B op, k=s contiguous)
  __shared__ __align__(16) bf16  sS   [64 * 64];   // [t][s]   masked scores (A op)
  __shared__ float sAprev[D_];

  const int tid = threadIdx.x, lane = tid & 31, wave = tid >> 5;
  const int bh = blockIdx.x, bidx = bh >> 4, hidx = bh & 15;
  const size_t base = (size_t)bh * (T_ * D_);

  for (int i = tid; i < D_ * D_; i += 256) sMt[i] = 0.0f;
  if (tid < D_) sAprev[tid] = 0.0f;
  __syncthreads();

  for (int chunk = 0; chunk < T_ / 64; ++chunk) {
    const size_t cb = base + (size_t)chunk * (64 * D_);
    // stage q (row-major) and b (transposed) as bf16
#pragma unroll
    for (int it = 0; it < 4; ++it) {
      const int i = it * 1024 + tid * 4;
      const int s = i >> 6, d = i & 63;
      float4 q4 = *(const float4*)(qln + cb + i);
      float4 b4 = *(const float4*)(bln + cb + i);
      sQ[i] = f2bf(q4.x); sQ[i+1] = f2bf(q4.y); sQ[i+2] = f2bf(q4.z); sQ[i+3] = f2bf(q4.w);
      sBt[(d + 0) * 64 + s] = f2bf(b4.x);
      sBt[(d + 1) * 64 + s] = f2bf(b4.y);
      sBt[(d + 2) * 64 + s] = f2bf(b4.z);
      sBt[(d + 3) * 64 + s] = f2bf(b4.w);
    }
    // exclusive cumsum along time (f32 carry across chunks), column-parallel;
    // write both layouts of aex
    if (tid < D_) {
      float run = sAprev[tid];
      for (int s = 0; s < 64; ++s) {
        bf16 v = f2bf(run);
        sAex [s * D_ + tid] = v;
        sAexT[tid * 64 + s] = v;
        run += aln[cb + s * D_ + tid];
      }
      sAprev[tid] = run;
    }
    __syncthreads();

    // phase 1 per wave (2 sub-tiles): o = q@M ; S = mask(q@aex^T) -> sS (bf16)
    v8f oacc[2];
#pragma unroll
    for (int i = 0; i < 2; ++i) {
      const int st = wave * 2 + i, tr = st >> 2, tc = st & 3;
      v8f acc = (v8f){0.f,0.f,0.f,0.f,0.f,0.f,0.f,0.f};
      v8f sc  = (v8f){0.f,0.f,0.f,0.f,0.f,0.f,0.f,0.f};
#pragma unroll
      for (int k0 = 0; k0 < D_; k0 += 32) {
        v16bf aq = frag_A(&sQ[(tr * 16) * D_ + k0], D_, lane);
        v16bf bm = frag_Bk_f32(sMt, D_, tc * 16, k0, lane);   // B[k][n] = M^T[n][k]
        acc = WMMA_BF16(aq, bm, acc);
        v16bf bt = frag_Bk(sAex, D_, tc * 16, k0, lane);      // B[k=d][n=s]=aex[s][d]
        sc = WMMA_BF16(aq, bt, sc);
      }
      oacc[i] = acc;
      const int n = tc * 16 + (lane & 15);
      const int mb = tr * 16 + ((lane >> 4) << 3);
#pragma unroll
      for (int r = 0; r < 8; ++r) {
        const int t = mb + r;
        sS[t * 64 + n] = f2bf((n <= t) ? sc[r] : 0.0f);       // causal: s <= t
      }
    }
    __syncthreads();

    // phase 2: o += S@b (store) ; M^T += (aex^T @ b)^T (state update)
#pragma unroll
    for (int i = 0; i < 2; ++i) {
      const int st = wave * 2 + i, tr = st >> 2, tc = st & 3;
      v8f acc = oacc[i];
      v8f macc = (v8f){0.f,0.f,0.f,0.f,0.f,0.f,0.f,0.f};
#pragma unroll
      for (int k0 = 0; k0 < D_; k0 += 32) {
        v16bf as = frag_A(&sS[(tr * 16) * 64 + k0], 64, lane);
        v16bf bb = frag_Bk(sBt, 64, tc * 16, k0, lane);       // B[k=s][n=d]=b[s][d]
        acc = WMMA_BF16(as, bb, acc);
        v16bf at = frag_A(&sAexT[(tr * 16) * 64 + k0], 64, lane); // A[m=d][k=s]
        macc = WMMA_BF16(at, bb, macc);
      }
      const int n = tc * 16 + (lane & 15);
      const int mb = tr * 16 + ((lane >> 4) << 3);
#pragma unroll
      for (int r = 0; r < 8; ++r) {
        const int t = chunk * 64 + mb + r;
        o[((size_t)(bidx * T_ + t)) * (H_ * D_) + hidx * D_ + n] = f2bf(acc[r]);
        sMt[n * D_ + (mb + r)] += macc[r];   // transposed store, contiguous in r
      }
    }
    __syncthreads();
  }
}

// ---------------------------------------------------------------------------
extern "C" void kernel_launch(void* const* d_in, const int* in_sizes, int n_in,
                              void* d_out, int out_size, void* d_ws, size_t ws_size,
                              hipStream_t stream) {
  const float* x   = (const float*)d_in[0];
  const float* Wq  = (const float*)d_in[1];
  const float* Wa  = (const float*)d_in[2];
  const float* Wb  = (const float*)d_in[3];
  const float* Wo  = (const float*)d_in[4];
  const float* qg  = (const float*)d_in[5];
  const float* qbt = (const float*)d_in[6];
  const float* ag  = (const float*)d_in[7];
  const float* abt = (const float*)d_in[8];
  const float* bg  = (const float*)d_in[9];
  const float* bbt = (const float*)d_in[10];
  float* out = (float*)d_out;

  char* p = (char*)d_ws;
  auto carve = [&](size_t bytes) {
    char* r = p; p += (bytes + 255) & ~(size_t)255; return r;
  };
  bf16*  xh   = (bf16*)carve((size_t)BT_ * DM_ * 2);
  bf16*  Wqt  = (bf16*)carve((size_t)DM_ * DM_ * 2);   // transposed bf16 weights
  bf16*  Wat  = (bf16*)carve((size_t)DM_ * DM_ * 2);
  bf16*  Wbt  = (bf16*)carve((size_t)DM_ * DM_ * 2);
  bf16*  Wot  = (bf16*)carve((size_t)DM_ * DM_ * 2);
  float* ysc  = (float*)carve((size_t)BT_ * DM_ * 4);
  float* qln  = (float*)carve((size_t)BT_ * DM_ * 4);
  float* alnp = (float*)carve((size_t)BT_ * DM_ * 4);
  float* blnp = (float*)carve((size_t)BT_ * DM_ * 4);
  bf16*  oh   = (bf16*)carve((size_t)BT_ * DM_ * 2);

  cvt_f32_bf16<<<dim3((BT_ * DM_ / 4 + 255) / 256), dim3(256), 0, stream>>>(x, xh, BT_ * DM_);
  const int wBlocks = (DM_ * DM_ + 255) / 256;
  cvt_f32_bf16_t<<<wBlocks, 256, 0, stream>>>(Wq, Wqt, DM_, DM_);
  cvt_f32_bf16_t<<<wBlocks, 256, 0, stream>>>(Wa, Wat, DM_, DM_);
  cvt_f32_bf16_t<<<wBlocks, 256, 0, stream>>>(Wb, Wbt, DM_, DM_);
  cvt_f32_bf16_t<<<wBlocks, 256, 0, stream>>>(Wo, Wot, DM_, DM_);

  const dim3 gg(DM_ / GTN, BT_ / GTM), gb(256);
  const int lnBlocks = (BT_ * H_) / 8;

  gemm_bf16f32<<<gg, gb, 0, stream>>>(xh, Wqt, ysc, BT_, DM_, DM_);
  ln_split<<<lnBlocks, 256, 0, stream>>>(ysc, qg, qbt, qln);

  gemm_bf16f32<<<gg, gb, 0, stream>>>(xh, Wat, ysc, BT_, DM_, DM_);
  ln_split<<<lnBlocks, 256, 0, stream>>>(ysc, ag, abt, alnp);

  gemm_bf16f32<<<gg, gb, 0, stream>>>(xh, Wbt, ysc, BT_, DM_, DM_);
  ln_split<<<lnBlocks, 256, 0, stream>>>(ysc, bg, bbt, blnp);

  attn_kernel<<<dim3(B_ * H_), dim3(256), 0, stream>>>(qln, alnp, blnp, oh);

  gemm_bf16f32<<<gg, gb, 0, stream>>>(oh, Wot, out, BT_, DM_, DM_);
}
